// RNNModel_82901458747988
// MI455X (gfx1250) — compile-verified, matrix operand order checked by reading, not run on hardware
//
#include <hip/hip_runtime.h>
#include <hip/hip_bf16.h>

typedef __attribute__((ext_vector_type(16))) __bf16 v16bf;
typedef __attribute__((ext_vector_type(8)))  __bf16 v8bf;
typedef __attribute__((ext_vector_type(8)))  float  v8f;

// ---------------------------------------------------------------------------
// Fragment loader: 16-bit WMMA operand layout (A 16x32 or B 32x16 transposed
// source [16 rows x K]). Lane r (0..15) half 0 holds K = k..k+7 and k+16..k+23;
// half 1 holds K = k+8..k+15 and k+24..k+31. Two 16B loads -> global_load_b128.
// ---------------------------------------------------------------------------
__device__ __forceinline__ v16bf load_frag16(const __bf16* __restrict__ base,
                                             int row, int ldk, int k, int half) {
    const __bf16* p = base + (long)row * ldk + k + half * 8;
    union { struct { v8bf lo; v8bf hi; } s; v16bf v; } u;
    u.s.lo = *(const v8bf*)(p);
    u.s.hi = *(const v8bf*)(p + 16);
    return u.v;
}

// ---------------------------------------------------------------------------
// pre[m,n] = sum_k A[m,k]*W[n,k] + b1[n] + b2[n]      (A:[M,K] bf16, W:[N,K] bf16)
// Each wave: one 16(M) x 64(N) strip, 4 accumulators, A-fragment reused 4x.
// ---------------------------------------------------------------------------
__global__ void __launch_bounds__(256)
gemm_pre_kernel(const __bf16* __restrict__ A, const __bf16* __restrict__ W,
                const float* __restrict__ b1, const float* __restrict__ b2,
                float* __restrict__ out, int M, int N, int K) {
    const int wave = threadIdx.x >> 5;
    const int lane = threadIdx.x & 31;
    const int half = lane >> 4;
    const int r    = lane & 15;

    const int nStrips = N >> 6;                 // strips of 64 columns
    const int task    = blockIdx.x * 8 + wave;  // one strip per wave
    const int m0      = (task / nStrips) * 16;
    const int n0      = (task % nStrips) * 64;

    v8f acc[4] = {};
    for (int k = 0; k < K; k += 32) {
        v16bf a = load_frag16(A, m0 + r, K, k, half);
#pragma unroll
        for (int j = 0; j < 4; ++j) {
            v16bf b = load_frag16(W, n0 + j * 16 + r, K, k, half);
            acc[j] = __builtin_amdgcn_wmma_f32_16x16x32_bf16(
                false, a, false, b, (short)0, acc[j], false, false);
        }
    }
    // f32 C/D layout: VGPR e -> row m0 + e + half*8, col n0 + 16j + r
    const int rowOff = half * 8;
#pragma unroll
    for (int j = 0; j < 4; ++j) {
        const int n = n0 + j * 16 + r;
        const float bias = b1[n] + b2[n];
#pragma unroll
        for (int e = 0; e < 8; ++e) {
            const long m = m0 + rowOff + e;
            out[m * (long)N + n] = acc[j][e] + bias;
        }
    }
}

// ---------------------------------------------------------------------------
// One recurrence step:  h_out = tanh(pre[:,t,:] + h_in @ Whh^T)
// M=B=64, N=K=H=1024 -> 64 wave-tasks -> 8 blocks x 8 waves.
// ---------------------------------------------------------------------------
__global__ void __launch_bounds__(256)
rnn_step_kernel(const __bf16* __restrict__ h_in,   // [B,H] bf16
                const __bf16* __restrict__ Whh,    // [H,H] bf16 (row n holds k)
                const float*  __restrict__ pre,    // [B,T,H] f32
                __bf16* __restrict__ h_out,        // [B,H] bf16
                __bf16* __restrict__ seq_out,      // [B,T,H] bf16
                int t, int T, int H) {
    const int wave = threadIdx.x >> 5;
    const int lane = threadIdx.x & 31;
    const int half = lane >> 4;
    const int r    = lane & 15;

    const int nStrips = H >> 6;
    const int task    = blockIdx.x * 8 + wave;
    const int m0      = (task / nStrips) * 16;   // batch tile
    const int n0      = (task % nStrips) * 64;   // hidden tile

    v8f acc[4] = {};
    for (int k = 0; k < H; k += 32) {
        v16bf a = load_frag16(h_in, m0 + r, H, k, half);
#pragma unroll
        for (int j = 0; j < 4; ++j) {
            v16bf b = load_frag16(Whh, n0 + j * 16 + r, H, k, half);
            acc[j] = __builtin_amdgcn_wmma_f32_16x16x32_bf16(
                false, a, false, b, (short)0, acc[j], false, false);
        }
    }
    const int rowOff = half * 8;
#pragma unroll
    for (int j = 0; j < 4; ++j) {
        const int n = n0 + j * 16 + r;
#pragma unroll
        for (int e = 0; e < 8; ++e) {
            const int  b   = m0 + rowOff + e;
            const long pix = ((long)b * T + t) * H + n;
            const float v  = tanhf(pre[pix] + acc[j][e]);
            const __bf16 hv = (__bf16)v;
            h_out[b * H + n] = hv;
            seq_out[pix]     = hv;
        }
    }
}

// ---------------------------------------------------------------------------
__global__ void cvt_f32_bf16_kernel(const float* __restrict__ in,
                                    __bf16* __restrict__ out, long n) {
    long i = (long)blockIdx.x * blockDim.x + threadIdx.x;
    if (i < n) out[i] = (__bf16)in[i];
}

__global__ void zero_bf16_kernel(__bf16* __restrict__ p, int n) {
    int i = blockIdx.x * blockDim.x + threadIdx.x;
    if (i < n) p[i] = (__bf16)0.0f;
}

// out[b,c] = h_last[b,:] . Wfc[c,:] + bfc[c]
__global__ void fc_kernel(const __bf16* __restrict__ seq, const float* __restrict__ Wfc,
                          const float* __restrict__ bfc, float* __restrict__ out,
                          int B, int T, int H, int C) {
    int idx = blockIdx.x * blockDim.x + threadIdx.x;
    if (idx >= B * C) return;
    int b = idx / C, c = idx % C;
    const __bf16* h = seq + ((long)b * T + (T - 1)) * H;
    float s = bfc[c];
    for (int k = 0; k < H; ++k) s += (float)h[k] * Wfc[c * H + k];
    out[idx] = s;
}

// ---------------------------------------------------------------------------
extern "C" void kernel_launch(void* const* d_in, const int* in_sizes, int n_in,
                              void* d_out, int out_size, void* d_ws, size_t ws_size,
                              hipStream_t stream) {
    constexpr int B = 64, T = 512, I = 256, H = 1024, C = 10;

    const float* x    = (const float*)d_in[0];
    const float* Wih0 = (const float*)d_in[1];
    const float* Whh0 = (const float*)d_in[2];
    const float* bih0 = (const float*)d_in[3];
    const float* bhh0 = (const float*)d_in[4];
    const float* Wih1 = (const float*)d_in[5];
    const float* Whh1 = (const float*)d_in[6];
    const float* bih1 = (const float*)d_in[7];
    const float* bhh1 = (const float*)d_in[8];
    const float* Wfc  = (const float*)d_in[9];
    const float* bfc  = (const float*)d_in[10];
    float* out = (float*)d_out;

    // --- workspace partition (256B aligned slices) ---
    char* w = (char*)d_ws;
    auto take = [&](size_t bytes) {
        char* p = w;
        w += (bytes + 255) & ~size_t(255);
        return p;
    };
    __bf16* x_bf    = (__bf16*)take((size_t)B * T * I * 2);   // 16 MB
    __bf16* Wih0_bf = (__bf16*)take((size_t)H * I * 2);       // 0.5 MB
    __bf16* Whh0_bf = (__bf16*)take((size_t)H * H * 2);       // 2 MB
    __bf16* Wih1_bf = (__bf16*)take((size_t)H * H * 2);       // 2 MB
    __bf16* Whh1_bf = (__bf16*)take((size_t)H * H * 2);       // 2 MB
    float*  pre     = (float*) take((size_t)B * T * H * 4);   // 128 MB
    __bf16* seq     = (__bf16*)take((size_t)B * T * H * 2);   // 64 MB (reused for both layers)
    __bf16* hA      = (__bf16*)take((size_t)B * H * 2);
    __bf16* hB      = (__bf16*)take((size_t)B * H * 2);

    // --- bf16 conversions ---
    {
        long n = (long)B * T * I;
        cvt_f32_bf16_kernel<<<(n + 255) / 256, 256, 0, stream>>>(x, x_bf, n);
        n = (long)H * I;
        cvt_f32_bf16_kernel<<<(n + 255) / 256, 256, 0, stream>>>(Wih0, Wih0_bf, n);
        n = (long)H * H;
        cvt_f32_bf16_kernel<<<(n + 255) / 256, 256, 0, stream>>>(Whh0, Whh0_bf, n);
        cvt_f32_bf16_kernel<<<(n + 255) / 256, 256, 0, stream>>>(Wih1, Wih1_bf, n);
        cvt_f32_bf16_kernel<<<(n + 255) / 256, 256, 0, stream>>>(Whh1, Whh1_bf, n);
    }

    const int M = B * T;                         // 32768
    const int gemmBlocks = (M / 16) * (H / 64) / 8;   // 4096
    const int stepBlocks = (B / 16) * (H / 64) / 8;   // 8

    // ================= layer 0 =================
    gemm_pre_kernel<<<gemmBlocks, 256, 0, stream>>>(x_bf, Wih0_bf, bih0, bhh0, pre, M, H, I);
    zero_bf16_kernel<<<(B * H + 255) / 256, 256, 0, stream>>>(hA, B * H);
    for (int t = 0; t < T; ++t) {
        const __bf16* hin  = (t & 1) ? hB : hA;
        __bf16*       hout = (t & 1) ? hA : hB;
        rnn_step_kernel<<<stepBlocks, 256, 0, stream>>>(hin, Whh0_bf, pre, hout, seq, t, T, H);
    }

    // ================= layer 1 =================
    gemm_pre_kernel<<<gemmBlocks, 256, 0, stream>>>(seq, Wih1_bf, bih1, bhh1, pre, M, H, H);
    zero_bf16_kernel<<<(B * H + 255) / 256, 256, 0, stream>>>(hA, B * H);
    for (int t = 0; t < T; ++t) {
        const __bf16* hin  = (t & 1) ? hB : hA;
        __bf16*       hout = (t & 1) ? hA : hB;
        rnn_step_kernel<<<stepBlocks, 256, 0, stream>>>(hin, Whh1_bf, pre, hout, seq, t, T, H);
    }

    // ================= FC head =================
    fc_kernel<<<(B * C + 255) / 256, 256, 0, stream>>>(seq, Wfc, bfc, out, B, T, H, C);
}
